// Formula_Encoding_24094766531308
// MI455X (gfx1250) — matrix operand order repaired and checked by reading, not compiled, not run on hardware
//
#include <hip/hip_runtime.h>

typedef __attribute__((ext_vector_type(16))) __bf16 v16bf;
typedef __attribute__((ext_vector_type(8)))  float  v8f;

struct Bytes32 { uint4 a, b; };   // 32 bytes, bit-cast target for v16bf

__device__ __forceinline__ unsigned short f2bf(float f) {
    unsigned u = __builtin_bit_cast(unsigned, f);
    u = (u + 0x7FFFu + ((u >> 16) & 1u)) >> 16;   // RTNE
    return (unsigned short)u;
}
__device__ __forceinline__ unsigned pk2(float lo, float hi) {
    return (unsigned)f2bf(lo) | ((unsigned)f2bf(hi) << 16);
}

// ---------------------------------------------------------------------------
// Prep: repack Wm/Wg (f32, row-major (128,384)) into bf16 B-fragments.
// Fragment order: [mat][ntile*12+ktile][lane][i], 512 bf16 = 1KB per fragment.
// B element for lane l, elem i:  n = ntile*16 + (l&15),  k = ktile*32 + (l>>4)*8 + (i&7) + (i>>3)*16
// B[k][n] = W[n][k]  (since GEMM is x @ W.T).
// ---------------------------------------------------------------------------
__global__ void prep_weights(const float* __restrict__ Wm,
                             const float* __restrict__ Wg,
                             unsigned short* __restrict__ ws) {
    int flat = blockIdx.x * 256 + threadIdx.x;        // 0 .. 98303
    int mat  = flat / 49152;
    int rem  = flat - mat * 49152;
    int frag = rem >> 9;                              // ntile*12 + ktile
    int lanei = rem & 511;
    int lane  = lanei >> 4;
    int i     = lanei & 15;
    int ntile = frag / 12;
    int ktile = frag - ntile * 12;
    int n = ntile * 16 + (lane & 15);
    int k = ktile * 32 + ((lane >> 4) << 3) + (i & 7) + ((i >> 3) << 4);
    const float* W = mat ? Wg : Wm;
    ws[flat] = f2bf(W[n * 384 + k]);
}

// ---------------------------------------------------------------------------
// Persistent scan kernel: 512 blocks x 64 threads (2 waves), 64 rows/block,
// 32 rows (2 M-tiles) per wave. 17 merge steps fully on-chip.
// ---------------------------------------------------------------------------
#define ROWS   64
#define XPITCH 392   // bf16 elems per row: 384 + 8 pad (row stride 784B = 49*16)

__global__ __launch_bounds__(64, 4)
void formula_scan(const float* __restrict__ ent_table,
                  const float* __restrict__ op_table,
                  const float* __restrict__ bmv_g,
                  const float* __restrict__ bgv_g,
                  const int*   __restrict__ ops_idx,
                  const int*   __restrict__ ents_idx,
                  const int*   __restrict__ left_idx,
                  const unsigned short* __restrict__ wfrag,
                  float* __restrict__ out) {
    __shared__ __align__(16) unsigned short xlds[ROWS * XPITCH];

    const int tid  = threadIdx.x;
    const int lane = tid & 31;
    const int wave = tid >> 5;
    const int blockRow0 = blockIdx.x * ROWS;
    const int kh = lane >> 4;        // K-half selector
    const int ml = lane & 15;        // M / N within tile

    // ---- stage acc0 = ent_table[ents_idx[:,16]] into x[256..384) ----
    {
        int r = tid;
        int grow = blockRow0 + r;
        int eidx = ents_idx[grow * 17 + 16];
        const float4* erow = (const float4*)(ent_table + (long)eidx * 128);
        unsigned* dst = (unsigned*)&xlds[r * XPITCH + 256];
        #pragma unroll 4
        for (int c4 = 0; c4 < 32; ++c4) {
            float4 v = erow[c4];
            uint2 p; p.x = pk2(v.x, v.y); p.y = pk2(v.z, v.w);
            *(uint2*)&dst[c4 * 2] = p;
        }
    }

    for (int s = 0; s < 17; ++s) {
        // ---------------- stage o,e for this step ----------------
        {
            int r = tid;
            int grow = blockRow0 + r;
            int oidx, eidx;
            if (s < 16) {
                int j = 15 - s;                       // flipped scan order
                oidx = ops_idx[grow * 16 + j];
                eidx = ents_idx[grow * 17 + j];
            } else {
                oidx = 4;                              // eq_e = op_table[4]
                eidx = left_idx[grow];
            }
            const float4* orow = (const float4*)(op_table  + (long)oidx * 128);
            const float4* erow = (const float4*)(ent_table + (long)eidx * 128);
            unsigned* dsto = (unsigned*)&xlds[r * XPITCH];
            unsigned* dste = (unsigned*)&xlds[r * XPITCH + 128];
            #pragma unroll 4
            for (int c4 = 0; c4 < 32; ++c4) {
                float4 ov = orow[c4];
                uint2 po; po.x = pk2(ov.x, ov.y); po.y = pk2(ov.z, ov.w);
                *(uint2*)&dsto[c4 * 2] = po;
                float4 ev = erow[c4];
                uint2 pe; pe.x = pk2(ev.x, ev.y); pe.y = pk2(ev.z, ev.w);
                *(uint2*)&dste[c4 * 2] = pe;
            }

            // prefetch next step's entity row into WGP$/L2 while GEMM runs
            if (s < 16) {
                int eidx2 = (s < 15) ? ents_idx[grow * 17 + (14 - s)]
                                     : left_idx[grow];
                const char* p = (const char*)(ent_table + (long)eidx2 * 128);
                __builtin_prefetch(p,       0, 3);
                __builtin_prefetch(p + 128, 0, 3);
                __builtin_prefetch(p + 256, 0, 3);
                __builtin_prefetch(p + 384, 0, 3);
            }
        }
        __syncthreads();

        // ---------------- GEMM + activation ----------------
        const int rowA0 = wave * 32 + ml;          // A-read row, M-tile 0
        const int rowA1 = rowA0 + 16;              // A-read row, M-tile 1
        unsigned res[8][8];                        // packed bf16 new-acc (s<16)

        #pragma unroll
        for (int nt = 0; nt < 8; ++nt) {
            // Anti-LICM barrier on a zero SGPR offset: keeps the weight loads
            // inside the scan loop (no hoist->spill) while preserving the
            // global address-space provenance (global_load_b128, not flat).
            int zoff = 0;
            asm volatile("" : "+s"(zoff));
            const Bytes32* wb = (const Bytes32*)wfrag + zoff;

            int col = nt * 16 + ml;
            float bmv = bmv_g[col];
            float bgv = bgv_g[col];
            v8f accm0 = {bmv,bmv,bmv,bmv,bmv,bmv,bmv,bmv};
            v8f accm1 = accm0;
            v8f accg0 = {bgv,bgv,bgv,bgv,bgv,bgv,bgv,bgv};
            v8f accg1 = accg0;

            for (int kt = 0; kt < 12; ++kt) {
                int fm = nt * 12 + kt;             // Wm fragment
                int fg = fm + 96;                  // Wg fragment
                v16bf Bm = __builtin_bit_cast(v16bf, wb[fm * 32 + lane]);
                v16bf Bg = __builtin_bit_cast(v16bf, wb[fg * 32 + lane]);

                int acol = kt * 32 + kh * 8;       // A layout: i<8 -> k=32kt+8kh+i, i>=8 -> +16
                Bytes32 a0, a1;
                a0.a = *(const uint4*)&xlds[rowA0 * XPITCH + acol];
                a0.b = *(const uint4*)&xlds[rowA0 * XPITCH + acol + 16];
                a1.a = *(const uint4*)&xlds[rowA1 * XPITCH + acol];
                a1.b = *(const uint4*)&xlds[rowA1 * XPITCH + acol + 16];
                v16bf A0 = __builtin_bit_cast(v16bf, a0);
                v16bf A1 = __builtin_bit_cast(v16bf, a1);

                accm0 = __builtin_amdgcn_wmma_f32_16x16x32_bf16(false, A0, false, Bm, (short)0, accm0, false, false);
                accm1 = __builtin_amdgcn_wmma_f32_16x16x32_bf16(false, A1, false, Bm, (short)0, accm1, false, false);
                accg0 = __builtin_amdgcn_wmma_f32_16x16x32_bf16(false, A0, false, Bg, (short)0, accg0, false, false);
                accg1 = __builtin_amdgcn_wmma_f32_16x16x32_bf16(false, A1, false, Bg, (short)0, accg1, false, false);
            }

            // epilogue: tanh(m) * sigmoid(g). C/D layout: VGPR v -> M = v + 8*kh.
            #pragma unroll
            for (int v = 0; v < 8; ++v) {
                float v0 = tanhf(accm0[v]) / (1.0f + __expf(-accg0[v]));
                float v1 = tanhf(accm1[v]) / (1.0f + __expf(-accg1[v]));
                if (s == 16) {
                    int m  = v + kh * 8;
                    int r0 = wave * 32 + m;
                    out[(long)(blockRow0 + r0) * 128 + col]      = v0;
                    out[(long)(blockRow0 + r0 + 16) * 128 + col] = v1;
                } else {
                    res[nt][v] = pk2(v0, v1);   // defer: kt>=8 of later nt still reads old acc
                }
            }
        }

        // write new acc back to x[256..384) after all reads of old acc finished
        if (s < 16) {
            #pragma unroll
            for (int nt = 0; nt < 8; ++nt) {
                #pragma unroll
                for (int v = 0; v < 8; ++v) {
                    unsigned u = res[nt][v];
                    int m   = v + kh * 8;
                    int col = nt * 16 + ml;
                    xlds[(wave * 32 + m)      * XPITCH + 256 + col] = (unsigned short)(u & 0xFFFFu);
                    xlds[(wave * 32 + m + 16) * XPITCH + 256 + col] = (unsigned short)(u >> 16);
                }
            }
        }
        __syncthreads();
    }
}

__global__ void copy_non(const float* __restrict__ non, float* __restrict__ out) {
    out[(long)32768 * 128 + threadIdx.x] = non[threadIdx.x];
}

extern "C" void kernel_launch(void* const* d_in, const int* in_sizes, int n_in,
                              void* d_out, int out_size, void* d_ws, size_t ws_size,
                              hipStream_t stream) {
    (void)in_sizes; (void)n_in; (void)out_size; (void)ws_size;
    const float* ent = (const float*)d_in[0];
    const float* opt = (const float*)d_in[1];
    const float* non = (const float*)d_in[2];
    const float* Wm  = (const float*)d_in[3];
    const float* bm  = (const float*)d_in[4];
    const float* Wg  = (const float*)d_in[5];
    const float* bg  = (const float*)d_in[6];
    const int* ops_idx  = (const int*)d_in[7];
    const int* ents_idx = (const int*)d_in[8];
    const int* left_idx = (const int*)d_in[9];
    float* out = (float*)d_out;
    unsigned short* wsb = (unsigned short*)d_ws;   // 192 KB bf16 weight fragments

    prep_weights<<<384, 256, 0, stream>>>(Wm, Wg, wsb);
    formula_scan<<<512, 64, 0, stream>>>(ent, opt, bm, bg,
                                         ops_idx, ents_idx, left_idx, wsb, out);
    copy_non<<<1, 128, 0, stream>>>(non, out);
}
